// GRN_48309792146002
// MI455X (gfx1250) — matrix operand order repaired and checked by reading, not compiled, not run on hardware
//
#include <hip/hip_runtime.h>
#include <hip/hip_bf16.h>

typedef _Float16 v16h __attribute__((ext_vector_type(16)));
typedef float    v8f  __attribute__((ext_vector_type(8)));

// ---------------------------------------------------------------------------
// WMMA fragment index maps (CDNA5 ISA 7.12.2, wave32, 16x16x32 f16):
//  A (16x32, MxK): lane m = L&15, hi = L>>4; element e -> K = (e<8?e:e+8)+8*hi
//  B (32x16, KxN): lane n = L&15, hi = L>>4; element e -> K = (e<8?e:e+8)+8*hi
//  C/D (16x16 f32): lane n = L&15; vgpr r -> M = r + 8*(L>>4)
// ---------------------------------------------------------------------------
__device__ __forceinline__ int kmapA(int e, int hi) { return ((e < 8) ? e : e + 8) + 8 * hi; }
__device__ __forceinline__ int kmapB(int e, int hi) { return ((e < 8) ? e : e + 8) + 8 * hi; }

__device__ __forceinline__ float masked_tap(const float* img, int H, int W, int wSh,
                                            int yi, int xi) {
    bool valid = (xi >= 0) && (xi <= W - 1) && (yi >= 0) && (yi <= H - 1);
    int yc = yi < 0 ? 0 : (yi > H - 1 ? H - 1 : yi);
    int xc = xi < 0 ? 0 : (xi > W - 1 ? W - 1 : xi);
    return valid ? img[((size_t)yc << wSh) + xc] : 0.0f;
}

// ===========================================================================
// Generic NCHW conv as implicit GEMM on WMMA.  K templated (const div/mod);
// all spatial sizes are powers of two -> shifts/masks only in the hot loop.
// Block = 128 threads = 4 waves, one 16(cout) x 16(pixel) tile per wave.
// ===========================================================================
template <int K>
__global__ __launch_bounds__(128) void conv2d_wmma_kernel(
    const float* __restrict__ X, const float* __restrict__ Wt,
    const float* __restrict__ Bs, float* __restrict__ Y,
    int Cin, int hSh, int wSh,        // input H = 1<<hSh, W = 1<<wSh
    int Cout, int stride, int pad,
    int hoSh, int woSh,               // output Ho = 1<<hoSh, Wo = 1<<woSh
    int act)
{
    __shared__ _Float16 sB[4][32 * 16];
    const int wave = threadIdx.x >> 5;
    const int lane = threadIdx.x & 31;
    const int m    = lane & 15;
    const int hi   = lane >> 4;
    const int p0   = (blockIdx.x * 4 + wave) * 16;
    const int co0  = blockIdx.y * 16;
    const int n    = blockIdx.z;
    const int k2   = K * K;
    const int Kt   = Cin * k2;
    const int H    = 1 << hSh, W = 1 << wSh;
    const int HWo  = 1 << (hoSh + woSh);
    const int woM  = (1 << woSh) - 1;
    const float* Xn = X + ((size_t)n * Cin << (hSh + wSh));

    v8f acc = {0.f, 0.f, 0.f, 0.f, 0.f, 0.f, 0.f, 0.f};

    for (int kb = 0; kb < Kt; kb += 32) {
        // ---- build 32(K) x 16(pixel) B tile in LDS (im2col, zero-padded)
#pragma unroll
        for (int e = 0; e < 16; ++e) {
            int q = lane * 16 + e;
            int kl = q >> 4, col = q & 15;
            int kk = kb + kl;
            float v = 0.f;
            int p = p0 + col;
            if (kk < Kt && p < HWo) {
                int c = kk / k2, t = kk % k2;          // compile-time divisor
                int oy = p >> woSh, ox = p & woM;
                int iy = oy * stride - pad + t / K;
                int ix = ox * stride - pad + t % K;
                if (iy >= 0 && iy < H && ix >= 0 && ix < W)
                    v = Xn[((size_t)c << (hSh + wSh)) + ((size_t)iy << wSh) + ix];
            }
            sB[wave][q] = (_Float16)v;
        }
        __syncthreads();

        // ---- A fragment straight from (L2-resident) weights; B from LDS
        v16h a, b;
        int co = co0 + m;
        if (kb + 32 < Kt) __builtin_prefetch(Wt + (size_t)co * Kt + kb + 32, 0, 3);
#pragma unroll
        for (int e = 0; e < 16; ++e) {
            int ka = kb + kmapA(e, hi);
            float av = (co < Cout && ka < Kt) ? Wt[(size_t)co * Kt + ka] : 0.f;
            a[e] = (_Float16)av;
            b[e] = sB[wave][kmapB(e, hi) * 16 + m];
        }
        acc = __builtin_amdgcn_wmma_f32_16x16x32_f16(false, a, false, b,
                                                     (short)0, acc, false, false);
        __syncthreads();
    }

#pragma unroll
    for (int r = 0; r < 8; ++r) {
        int co = co0 + r + 8 * hi;
        int p  = p0 + m;
        if (co < Cout && p < HWo) {
            float v = acc[r] + Bs[co];
            if (act == 1) v = fmaxf(v, 0.f);
            Y[(((size_t)n * Cout + co) << (hoSh + woSh)) + p] = v;
        }
    }
}

// ===========================================================================
// Deformable conv contraction on WMMA: out[o,p] = sum_{t,c} samp[c,t,p]*w[o,c,t]
// B tile = bilinear-gathered samples built on the fly (K index = t*Cin + c).
// stride 1, pad K/2, Ho==H, Wo==W; Cin power of two (cSh), H/W powers of two.
// ===========================================================================
template <int K>
__global__ __launch_bounds__(128) void deform_conv_wmma_kernel(
    const float* __restrict__ X, const float* __restrict__ Off,
    const float* __restrict__ Wt, float* __restrict__ Y,
    int cSh, int hSh, int wSh, int Cout, int act)
{
    __shared__ _Float16 sB[4][32 * 16];
    const int wave = threadIdx.x >> 5;
    const int lane = threadIdx.x & 31;
    const int m    = lane & 15;
    const int hi   = lane >> 4;
    const int p0   = (blockIdx.x * 4 + wave) * 16;
    const int co0  = blockIdx.y * 16;
    const int n    = blockIdx.z;
    const int k2   = K * K;
    const int Cin  = 1 << cSh;
    const int cM   = Cin - 1;
    const int Kt   = k2 << cSh;
    const int pad  = K / 2;
    const int H    = 1 << hSh, W = 1 << wSh;
    const int hwSh = hSh + wSh;
    const int HW   = 1 << hwSh;
    const int wM   = W - 1;
    const float* Xn   = X + ((size_t)n << (cSh + hwSh));
    const float* Offn = Off + (((size_t)n * (2 * k2)) << hwSh);

    v8f acc = {0.f, 0.f, 0.f, 0.f, 0.f, 0.f, 0.f, 0.f};

    for (int kb = 0; kb < Kt; kb += 32) {
#pragma unroll
        for (int e = 0; e < 16; ++e) {
            int q = lane * 16 + e;
            int kl = q >> 4, col = q & 15;
            int kk = kb + kl;
            float v = 0.f;
            int p = p0 + col;
            if (kk < Kt && p < HW) {
                int t = kk >> cSh, c = kk & cM;
                int oy = p >> wSh, ox = p & wM;
                float dy = Offn[((size_t)(2 * t) << hwSh) + p];
                float dx = Offn[((size_t)(2 * t + 1) << hwSh) + p];
                float fy = (float)(t / K - pad + oy) + dy;   // const divisor
                float fx = (float)(t % K - pad + ox) + dx;
                float y0f = floorf(fy), x0f = floorf(fx);
                int y0 = (int)y0f, x0 = (int)x0f;
                float wy = fy - y0f, wx = fx - x0f;
                const float* Xc = Xn + ((size_t)c << hwSh);
                v  = masked_tap(Xc, H, W, wSh, y0,     x0    ) * (1.f - wy) * (1.f - wx);
                v += masked_tap(Xc, H, W, wSh, y0,     x0 + 1) * (1.f - wy) * wx;
                v += masked_tap(Xc, H, W, wSh, y0 + 1, x0    ) * wy * (1.f - wx);
                v += masked_tap(Xc, H, W, wSh, y0 + 1, x0 + 1) * wy * wx;
            }
            sB[wave][q] = (_Float16)v;
        }
        __syncthreads();

        v16h a, b;
        int co = co0 + m;
#pragma unroll
        for (int e = 0; e < 16; ++e) {
            int ka = kb + kmapA(e, hi);
            float av = 0.f;
            if (co < Cout && ka < Kt) {
                int t = ka >> cSh, c = ka & cM;
                av = Wt[(((size_t)co << cSh) + c) * k2 + t];
            }
            a[e] = (_Float16)av;
            b[e] = sB[wave][kmapB(e, hi) * 16 + m];
        }
        acc = __builtin_amdgcn_wmma_f32_16x16x32_f16(false, a, false, b,
                                                     (short)0, acc, false, false);
        __syncthreads();
    }

#pragma unroll
    for (int r = 0; r < 8; ++r) {
        int co = co0 + r + 8 * hi;
        int p  = p0 + m;
        if (co < Cout && p < HW) {
            float v = acc[r];
            if (act == 1) v = fmaxf(v, 0.f);
            Y[(((size_t)n * Cout + co) << hwSh) + p] = v;
        }
    }
}

// ===========================================================================
// Non-local block, flash-attention style.  One wave = 16 query rows.
// S(16 x 32) = th^T x ph via 2 WMMAs; online softmax in LDS; out accumulated
// as P(16x32) x g^T(32x32) via 2 WMMAs.  Never materializes 4096x4096 logits.
// ===========================================================================
__global__ __launch_bounds__(32) void nl_attn_kernel(
    const float* __restrict__ TH, const float* __restrict__ PH,
    const float* __restrict__ G, float* __restrict__ OUT, int HW)
{
    const int IC = 32;
    __shared__ float     sS[16 * 32];
    __shared__ _Float16  sP[16 * 32];
    __shared__ float     sM[16], sL[16], sScale[16];
    const int lane = threadIdx.x;
    const int m    = lane & 15;
    const int hi   = lane >> 4;
    const int i0   = blockIdx.x * 16;
    const int n    = blockIdx.y;
    const float* th = TH + (size_t)n * IC * HW;
    const float* ph = PH + (size_t)n * IC * HW;
    const float* g  = G  + (size_t)n * IC * HW;

    if (lane < 16) { sM[lane] = -1e30f; sL[lane] = 0.f; }
    __syncthreads();

    v16h aTH;
#pragma unroll
    for (int e = 0; e < 16; ++e)
        aTH[e] = (_Float16)th[(size_t)kmapA(e, hi) * HW + i0 + m];

    v8f acc0 = {0.f,0.f,0.f,0.f,0.f,0.f,0.f,0.f};
    v8f acc1 = {0.f,0.f,0.f,0.f,0.f,0.f,0.f,0.f};
    const float scl = 0.17677669529663687f;   // 1/sqrt(32)

    for (int jb = 0; jb < HW; jb += 32) {
        v16h b0, b1;
#pragma unroll
        for (int e = 0; e < 16; ++e) {
            int c = kmapB(e, hi);
            b0[e] = (_Float16)ph[(size_t)c * HW + jb + m];
            b1[e] = (_Float16)ph[(size_t)c * HW + jb + 16 + m];
        }
        v8f z = {0.f,0.f,0.f,0.f,0.f,0.f,0.f,0.f};
        v8f s0 = __builtin_amdgcn_wmma_f32_16x16x32_f16(false, aTH, false, b0, (short)0, z, false, false);
        v8f s1 = __builtin_amdgcn_wmma_f32_16x16x32_f16(false, aTH, false, b1, (short)0, z, false, false);
#pragma unroll
        for (int r = 0; r < 8; ++r) {
            int row = r + 8 * hi;
            sS[row * 32 + m]      = s0[r] * scl;
            sS[row * 32 + 16 + m] = s1[r] * scl;
        }
        __syncthreads();

        if (lane < 16) {
            float mo = sM[lane];
            float rmax = mo;
            for (int j = 0; j < 32; ++j) rmax = fmaxf(rmax, sS[lane * 32 + j]);
            float sc = expf(mo - rmax);
            float ls = sL[lane] * sc;
            for (int j = 0; j < 32; ++j) {
                float p = expf(sS[lane * 32 + j] - rmax);
                ls += p;
                sP[lane * 32 + j] = (_Float16)p;
            }
            sM[lane] = rmax; sL[lane] = ls; sScale[lane] = sc;
        }
        __syncthreads();

#pragma unroll
        for (int r = 0; r < 8; ++r) {
            float sc = sScale[r + 8 * hi];
            acc0[r] *= sc; acc1[r] *= sc;
        }
        v16h aP, bg0, bg1;
#pragma unroll
        for (int e = 0; e < 16; ++e) {
            aP[e] = sP[m * 32 + kmapA(e, hi)];
            int kj = kmapB(e, hi);
            bg0[e] = (_Float16)g[(size_t)m * HW + jb + kj];
            bg1[e] = (_Float16)g[(size_t)(16 + m) * HW + jb + kj];
        }
        acc0 = __builtin_amdgcn_wmma_f32_16x16x32_f16(false, aP, false, bg0, (short)0, acc0, false, false);
        acc1 = __builtin_amdgcn_wmma_f32_16x16x32_f16(false, aP, false, bg1, (short)0, acc1, false, false);
        __syncthreads();
    }

#pragma unroll
    for (int r = 0; r < 8; ++r) {
        int row = r + 8 * hi;
        float inv = 1.f / sL[row];
        OUT[(size_t)n * IC * HW + (size_t)m * HW        + i0 + row] = acc0[r] * inv;
        OUT[(size_t)n * IC * HW + (size_t)(16 + m) * HW + i0 + row] = acc1[r] * inv;
    }
}

// ===========================================================================
// TPS warp: affine grid + TPS grid + grid_sample(ag, tg, align_corners=True).
// ag is analytic (affine of pixel coords) so the 4 taps are evaluated inline.
// ===========================================================================
__global__ void tps_warp_kernel(const float* __restrict__ Aff,
                                const float* __restrict__ Theta,
                                const float* __restrict__ Ctrl,
                                float* __restrict__ TPS, int N, int H, int W)
{
    int idx = blockIdx.x * blockDim.x + threadIdx.x;
    if (idx >= H * W) return;
    int y = idx / W, x = idx % W;
    float gx = (float)x / (float)(W - 1);
    float gy = (float)y / (float)(H - 1);

    float w0x = 0.f, w0y = 0.f;
    for (int t = 0; t < 35; ++t) { w0x -= Theta[2 * t]; w0y -= Theta[2 * t + 1]; }
    float zx = 0.f, zy = 0.f;
    for (int t = 0; t < 36; ++t) {
        float wxv = (t == 0) ? w0x : Theta[2 * (t - 1)];
        float wyv = (t == 0) ? w0y : Theta[2 * (t - 1) + 1];
        float dx = gx - Ctrl[2 * t], dy = gy - Ctrl[2 * t + 1];
        float D = sqrtf(dx * dx + dy * dy + 1e-12f);
        float U = D * D * logf(D + 1e-6f);
        zx += U * wxv; zy += U * wyv;
    }
    zx += Theta[2 * 35] + Theta[2 * 36] * gx + Theta[2 * 37] * gy;
    zy += Theta[2 * 35 + 1] + Theta[2 * 36 + 1] * gx + Theta[2 * 37 + 1] * gy;
    float tgx = (gx + zx) * 2.f - 1.f;
    float tgy = (gy + zy) * 2.f - 1.f;

    float xs = (tgx + 1.f) * 0.5f * (float)(W - 1);
    float ys = (tgy + 1.f) * 0.5f * (float)(H - 1);
    float x0f = floorf(xs), y0f = floorf(ys);
    int x0 = (int)x0f, y0 = (int)y0f;
    float wx = xs - x0f, wy = ys - y0f;

    for (int k = 0; k < 2; ++k) {
        float A0 = Aff[k * 3 + 0], A1 = Aff[k * 3 + 1], A2 = Aff[k * 3 + 2];
        auto agv = [&](int h, int w) -> float {
            float ax = (2.f * (float)w + 1.f) / (float)W - 1.f;
            float ay = (2.f * (float)h + 1.f) / (float)H - 1.f;
            return A0 * ax + A1 * ay + A2;
        };
        auto tapA = [&](int yi, int xi) -> float {
            bool valid = (xi >= 0) && (xi <= W - 1) && (yi >= 0) && (yi <= H - 1);
            int yc = yi < 0 ? 0 : (yi > H - 1 ? H - 1 : yi);
            int xc = xi < 0 ? 0 : (xi > W - 1 ? W - 1 : xi);
            return valid ? agv(yc, xc) : 0.f;
        };
        float v = tapA(y0, x0) * (1.f - wy) * (1.f - wx)
                + tapA(y0, x0 + 1) * (1.f - wy) * wx
                + tapA(y0 + 1, x0) * wy * (1.f - wx)
                + tapA(y0 + 1, x0 + 1) * wy * wx;
        for (int n = 0; n < N; ++n)
            TPS[(((size_t)n * 2 + k) * H + y) * W + x] = v;
    }
}

// elementwise: out = relu(a+b+c+d)
__global__ void combine4_relu_kernel(const float* a, const float* b, const float* c,
                                     const float* d, float* o, int nElem)
{
    int i = blockIdx.x * blockDim.x + threadIdx.x;
    if (i < nElem) o[i] = fmaxf(a[i] + b[i] + c[i] + d[i], 0.f);
}

// elementwise: out = relu(a+b+c+d+2h)   (nonlocal adds h, stage sum adds h again)
__global__ void combine5_relu_kernel(const float* a, const float* b, const float* c,
                                     const float* d, const float* h, float* o, int nElem)
{
    int i = blockIdx.x * blockDim.x + threadIdx.x;
    if (i < nElem) o[i] = fmaxf(a[i] + b[i] + c[i] + d[i] + 2.f * h[i], 0.f);
}

// 2x2 transposed conv (2x upsample): y[n,o,2h+i,2w+j] = sum_c x[n,c,h,w]*W[c,o,i,j]+b
__global__ void convt2_kernel(const float* __restrict__ X, const float* __restrict__ Wt,
                              const float* __restrict__ Bs, float* __restrict__ Y,
                              int N, int C, int O, int H, int W, int act)
{
    int H2 = 2 * H, W2 = 2 * W;
    int total = N * O * H2 * W2;
    int idx = blockIdx.x * blockDim.x + threadIdx.x;
    if (idx >= total) return;
    int x = idx % W2; int t = idx / W2;
    int y = t % H2;   t /= H2;
    int o = t % O;    int n = t / O;
    int h = y >> 1, i = y & 1, w = x >> 1, j = x & 1;
    float acc = Bs[o];
    for (int c = 0; c < C; ++c)
        acc += X[(((size_t)n * C + c) * H + h) * W + w] * Wt[(((size_t)c * O + o) * 2 + i) * 2 + j];
    if (act == 1) acc = fmaxf(acc, 0.f);
    Y[idx] = acc;
}

// leaky(t2) + tps -> clip -> grid_sample(x, fine, align_corners=False)
__global__ void final_warp_kernel(const float* __restrict__ Ximg,
                                  const float* __restrict__ T2,
                                  const float* __restrict__ TPS,
                                  float* __restrict__ OUT, int N, int C, int H, int W)
{
    int idx = blockIdx.x * blockDim.x + threadIdx.x;
    if (idx >= N * H * W) return;
    int x = idx % W, y = (idx / W) % H, n = idx / (W * H);
    float f[2];
    for (int ch = 0; ch < 2; ++ch) {
        float v = T2[(((size_t)n * 2 + ch) * H + y) * W + x];
        v = (v >= 0.f) ? v : 0.1f * v;
        v += TPS[(((size_t)n * 2 + ch) * H + y) * W + x];
        f[ch] = fminf(fmaxf(v, -1.f), 1.f);
    }
    float xs = ((f[0] + 1.f) * (float)W - 1.f) * 0.5f;
    float ys = ((f[1] + 1.f) * (float)H - 1.f) * 0.5f;
    float x0f = floorf(xs), y0f = floorf(ys);
    int x0 = (int)x0f, y0 = (int)y0f;
    float wx = xs - x0f, wy = ys - y0f;
    for (int c = 0; c < C; ++c) {
        const float* img = Ximg + ((size_t)n * C + c) * H * W;
        float v = masked_tap(img, H, W, 8, y0, x0) * (1.f - wy) * (1.f - wx)
                + masked_tap(img, H, W, 8, y0, x0 + 1) * (1.f - wy) * wx
                + masked_tap(img, H, W, 8, y0 + 1, x0) * wy * (1.f - wx)
                + masked_tap(img, H, W, 8, y0 + 1, x0 + 1) * wy * wx;
        OUT[(((size_t)n * C + c) * H + y) * W + x] = v;
    }
}

// ===========================================================================
// Host orchestration
// ===========================================================================
static int ilog2(int v) { int s = 0; while ((1 << s) < v) ++s; return s; }

static void launch_conv(const float* X, const float* Wt, const float* Bs, float* Y,
                        int N, int Cin, int H, int W, int Cout, int K, int s, int p,
                        int act, hipStream_t st)
{
    int Ho = (H + 2 * p - K) / s + 1;
    int Wo = (W + 2 * p - K) / s + 1;
    int hSh = ilog2(H), wSh = ilog2(W), hoSh = ilog2(Ho), woSh = ilog2(Wo);
    int pixTiles = (Ho * Wo + 15) / 16;
    dim3 grid((pixTiles + 3) / 4, (Cout + 15) / 16, N);
    switch (K) {
    case 1: conv2d_wmma_kernel<1><<<grid, 128, 0, st>>>(X, Wt, Bs, Y, Cin, hSh, wSh, Cout, s, p, hoSh, woSh, act); break;
    case 3: conv2d_wmma_kernel<3><<<grid, 128, 0, st>>>(X, Wt, Bs, Y, Cin, hSh, wSh, Cout, s, p, hoSh, woSh, act); break;
    case 5: conv2d_wmma_kernel<5><<<grid, 128, 0, st>>>(X, Wt, Bs, Y, Cin, hSh, wSh, Cout, s, p, hoSh, woSh, act); break;
    default: conv2d_wmma_kernel<7><<<grid, 128, 0, st>>>(X, Wt, Bs, Y, Cin, hSh, wSh, Cout, s, p, hoSh, woSh, act); break;
    }
}

static void launch_deform(const float* X, const float* Off, const float* Wt, float* Y,
                          int N, int Cin, int H, int W, int Cout, int K, int act,
                          hipStream_t st)
{
    int cSh = ilog2(Cin), hSh = ilog2(H), wSh = ilog2(W);
    int pixTiles = (H * W + 15) / 16;
    dim3 grid((pixTiles + 3) / 4, (Cout + 15) / 16, N);
    switch (K) {
    case 3: deform_conv_wmma_kernel<3><<<grid, 128, 0, st>>>(X, Off, Wt, Y, cSh, hSh, wSh, Cout, act); break;
    case 5: deform_conv_wmma_kernel<5><<<grid, 128, 0, st>>>(X, Off, Wt, Y, cSh, hSh, wSh, Cout, act); break;
    default: deform_conv_wmma_kernel<7><<<grid, 128, 0, st>>>(X, Off, Wt, Y, cSh, hSh, wSh, Cout, act); break;
    }
}

extern "C" void kernel_launch(void* const* d_in, const int* in_sizes, int n_in,
                              void* d_out, int out_size, void* d_ws, size_t ws_size,
                              hipStream_t stream)
{
    const int N = 2;
    const float* x_in  = (const float*)d_in[0];
    const float* aff   = (const float*)d_in[1];
    const float* theta = (const float*)d_in[2];
    const float* ctrl  = (const float*)d_in[3];
    const float* c1_w  = (const float*)d_in[4];
    const float* c1_b  = (const float*)d_in[5];
    // d1: k3 ow/ob/w = 6,7,8  k5 = 9,10,11  k7 = 12,13,14 ; d2: 15..23
    const float* c2_w  = (const float*)d_in[24];
    const float* c2_b  = (const float*)d_in[25];
    const float* n7_w  = (const float*)d_in[26]; const float* n7_b = (const float*)d_in[27];
    const float* n5_w  = (const float*)d_in[28]; const float* n5_b = (const float*)d_in[29];
    const float* n3_w  = (const float*)d_in[30]; const float* n3_b = (const float*)d_in[31];
    const float* g_w   = (const float*)d_in[32]; const float* g_b  = (const float*)d_in[33];
    const float* th_w  = (const float*)d_in[34]; const float* th_b = (const float*)d_in[35];
    const float* ph_w  = (const float*)d_in[36]; const float* ph_b = (const float*)d_in[37];
    const float* o_w   = (const float*)d_in[38]; const float* o_b  = (const float*)d_in[39];
    const float* t1_w  = (const float*)d_in[40]; const float* t1_b = (const float*)d_in[41];
    const float* t2_w  = (const float*)d_in[42]; const float* t2_b = (const float*)d_in[43];

    // ---- workspace bump allocator (floats), ~59 MB total
    char* wp = (char*)d_ws;
    auto alloc = [&](size_t nfloats) -> float* {
        float* p = (float*)wp;
        wp += (nfloats * sizeof(float) + 255) & ~(size_t)255;
        return p;
    };
    float* tpsb = alloc((size_t)N * 2 * 256 * 256);
    float* h1   = alloc((size_t)N * 32 * 128 * 128);
    float* offb = alloc((size_t)N * 98 * 128 * 128);   // reused for all offset convs
    float* bA   = alloc((size_t)N * 32 * 128 * 128);   // branch buffers (reused in dilm2)
    float* bB   = alloc((size_t)N * 32 * 128 * 128);
    float* bC   = alloc((size_t)N * 32 * 128 * 128);
    float* h1b  = alloc((size_t)N * 32 * 128 * 128);
    float* h2   = alloc((size_t)N * 64 * 64 * 64);
    float* h2b  = alloc((size_t)N * 64 * 64 * 64);
    float* x7b  = alloc((size_t)N * 64 * 64 * 64);
    float* x5b  = alloc((size_t)N * 64 * 64 * 64);
    float* x3b  = alloc((size_t)N * 64 * 64 * 64);
    float* thb  = alloc((size_t)N * 32 * 64 * 64);
    float* phb  = alloc((size_t)N * 32 * 64 * 64);
    float* gb   = alloc((size_t)N * 32 * 64 * 64);
    float* nlb  = alloc((size_t)N * 32 * 64 * 64);
    float* xnb  = alloc((size_t)N * 64 * 64 * 64);
    float* hcb  = alloc((size_t)N * 64 * 64 * 64);
    float* t1b  = alloc((size_t)N * 32 * 128 * 128);
    float* t2b  = alloc((size_t)N * 2 * 256 * 256);

    // 1) TPS warp (analytic affine grid sampled at TPS grid)
    tps_warp_kernel<<<(256 * 256 + 255) / 256, 256, 0, stream>>>(aff, theta, ctrl, tpsb, N, 256, 256);

    // 2) c1: 2 -> 32, k3 s2 p1 : 256 -> 128
    launch_conv(tpsb, c1_w, c1_b, h1, N, 2, 256, 256, 32, 3, 2, 1, 0, stream);

    // 3) dilm1 (32ch, 128x128)
    {
        const int ks[3] = {3, 5, 7};
        float* outs[3] = {bA, bB, bC};
        for (int kidx = 0; kidx < 3; ++kidx) {
            int k = ks[kidx];
            const float* ow = (const float*)d_in[6 + 3 * kidx + 0];
            const float* ob = (const float*)d_in[6 + 3 * kidx + 1];
            const float* ww = (const float*)d_in[6 + 3 * kidx + 2];
            launch_conv(h1, ow, ob, offb, N, 32, 128, 128, 2 * k * k, k, 1, k / 2, 0, stream);
            launch_deform(h1, offb, ww, outs[kidx], N, 32, 128, 128, 32, k, 1, stream);
        }
        int nE = N * 32 * 128 * 128;
        combine4_relu_kernel<<<(nE + 255) / 256, 256, 0, stream>>>(bA, bB, bC, h1, h1b, nE);
    }

    // 4) c2: 32 -> 64, k3 s2 p1 : 128 -> 64
    launch_conv(h1b, c2_w, c2_b, h2, N, 32, 128, 128, 64, 3, 2, 1, 0, stream);

    // 5) dilm2 (64ch, 64x64)
    {
        const int ks[3] = {3, 5, 7};
        float* outs[3] = {bA, bB, bC};
        for (int kidx = 0; kidx < 3; ++kidx) {
            int k = ks[kidx];
            const float* ow = (const float*)d_in[15 + 3 * kidx + 0];
            const float* ob = (const float*)d_in[15 + 3 * kidx + 1];
            const float* ww = (const float*)d_in[15 + 3 * kidx + 2];
            launch_conv(h2, ow, ob, offb, N, 64, 64, 64, 2 * k * k, k, 1, k / 2, 0, stream);
            launch_deform(h2, offb, ww, outs[kidx], N, 64, 64, 64, 64, k, 1, stream);
        }
        int nE = N * 64 * 64 * 64;
        combine4_relu_kernel<<<(nE + 255) / 256, 256, 0, stream>>>(bA, bB, bC, h2, h2b, nE);
    }

    // 6) multi-scale convs + nonlocal
    launch_conv(h2b, n7_w, n7_b, x7b, N, 64, 64, 64, 64, 7, 1, 3, 0, stream);
    launch_conv(h2b, n5_w, n5_b, x5b, N, 64, 64, 64, 64, 5, 1, 2, 0, stream);
    launch_conv(h2b, n3_w, n3_b, x3b, N, 64, 64, 64, 64, 3, 1, 1, 0, stream);
    launch_conv(h2b, g_w,  g_b,  gb,  N, 64, 64, 64, 32, 1, 1, 0, 0, stream);
    launch_conv(h2b, th_w, th_b, thb, N, 64, 64, 64, 32, 1, 1, 0, 0, stream);
    launch_conv(h2b, ph_w, ph_b, phb, N, 64, 64, 64, 32, 1, 1, 0, 0, stream);
    {
        dim3 grid(4096 / 16, N);
        nl_attn_kernel<<<grid, 32, 0, stream>>>(thb, phb, gb, nlb, 4096);
    }
    launch_conv(nlb, o_w, o_b, xnb, N, 32, 64, 64, 64, 1, 1, 0, 0, stream);
    {
        int nE = N * 64 * 64 * 64;   // h = relu(x7+x5+x3+(conv(nl)+h)+h)
        combine5_relu_kernel<<<(nE + 255) / 256, 256, 0, stream>>>(x7b, x5b, x3b, xnb, h2b, hcb, nE);
    }

    // 7) transposed 2x upsamples
    {
        int total1 = N * 32 * 128 * 128;
        convt2_kernel<<<(total1 + 255) / 256, 256, 0, stream>>>(hcb, t1_w, t1_b, t1b,
                                                                N, 64, 32, 64, 64, 1);
        int total2 = N * 2 * 256 * 256;
        convt2_kernel<<<(total2 + 255) / 256, 256, 0, stream>>>(t1b, t2_w, t2_b, t2b,
                                                                N, 32, 2, 128, 128, 0);
    }

    // 8) leaky + tps residual + clip + final grid_sample of the input image
    {
        int nPix = N * 256 * 256;
        final_warp_kernel<<<(nPix + 255) / 256, 256, 0, stream>>>(x_in, t2b, tpsb,
                                                                  (float*)d_out, N, 3, 256, 256);
    }
    (void)in_sizes; (void)n_in; (void)out_size; (void)ws_size;
}
